// CausalSelfAttention_17042430230786
// MI455X (gfx1250) — matrix operand order repaired and checked by reading, not compiled, not run on hardware
//
#include <hip/hip_runtime.h>

// MI455X / gfx1250, wave32. Fused QKV projection + full multi-head attention.
// Kernel 1: bf16 WMMA GEMM  [4096,1024]x[1024,3072]+bias -> bf16 Q/K/V in d_ws.
// Kernel 2: flash-attention (online softmax), bf16 WMMA for QK^T and PV,
//           async global->LDS staging (ASYNCcnt) with double-buffered K/V,
//           global_prefetch for the +2 chunk, fp32 output [T, N_HEADS, HEAD_DIM].
// d_ws usage: 3 * 16 * 4096 * 64 * 2 bytes = 24 MiB.

typedef __attribute__((ext_vector_type(16))) __bf16 v16bf;
typedef __attribute__((ext_vector_type(8)))  __bf16 v8bf;
typedef __attribute__((ext_vector_type(4)))  __bf16 v4bf;
typedef __attribute__((ext_vector_type(8)))  float  v8f;
typedef __attribute__((ext_vector_type(4)))  float  v4f;

#define T_SEQ 4096
#define EMB   1024
#define N3E   3072
#define NHEAD 16
#define HDIM  64

__device__ __forceinline__ __bf16 cvt_bf16(float f) {
  unsigned u = __builtin_bit_cast(unsigned, f);
  u += 0x7FFFu + ((u >> 16) & 1u);            // round-to-nearest-even
  unsigned short s = (unsigned short)(u >> 16);
  return __builtin_bit_cast(__bf16, s);
}

__device__ __forceinline__ v16bf concat8(v8bf lo, v8bf hi) {
  return __builtin_shufflevector(lo, hi, 0,1,2,3,4,5,6,7,8,9,10,11,12,13,14,15);
}

// low 32 bits of a generic pointer into LDS == wave-relative LDS byte offset
__device__ __forceinline__ unsigned lds_off(const void* p) {
  return (unsigned)(reinterpret_cast<uintptr_t>(p));
}

// async copy 16B global -> LDS (per-lane addresses), tracked by ASYNCcnt
__device__ __forceinline__ void async_copy_b128(unsigned lds_addr, const void* gptr) {
  unsigned long long ga = (unsigned long long)(uintptr_t)gptr;
  asm volatile("global_load_async_to_lds_b128 %0, %1, off"
               :: "v"(lds_addr), "v"(ga) : "memory");
}

__device__ __forceinline__ void wait_async0() {
  asm volatile("s_wait_asynccnt 0" ::: "memory");
}

// ---------------------------------------------------------------------------
// Kernel 1: QKV projection.  Block tile 128(M) x 128(N), K-steps of 32.
// 8 waves/block; wave (wm,wn) owns a 64x32 sub-tile = 4x2 wmma accumulators.
// ---------------------------------------------------------------------------
__global__ __launch_bounds__(256)
void qkv_gemm_bf16_wmma(const float* __restrict__ x,
                        const float* __restrict__ W,
                        const float* __restrict__ bias,
                        __bf16* __restrict__ qkv) {
  __shared__ __bf16 ldsA[128][40];   // [m][k], pad 40 (80B stride, 16B aligned)
  __shared__ __bf16 ldsB[128][40];   // [n][k] (transposed on store)

  const int tid  = threadIdx.x;
  const int lane = tid & 31;
  const int wave = tid >> 5;
  const int half = lane >> 4;        // 0..1
  const int l16  = lane & 15;        // 0..15
  const int mblk = blockIdx.y << 7;
  const int nblk = blockIdx.x << 7;
  const int wm   = (wave & 1) << 6;  // 0 / 64
  const int wn   = (wave >> 1) << 5; // 0 / 32 / 64 / 96

  v8f acc[4][2] = {};

  for (int k0 = 0; k0 < EMB; k0 += 32) {
    // ---- stage A: x[mblk..+127][k0..+31] fp32 -> bf16 LDS ----
#pragma unroll
    for (int i = 0; i < 4; ++i) {
      int idx = tid + (i << 8);             // 0..1023
      int row = idx >> 3;                   // 0..127
      int c   = (idx & 7) << 2;             // 0,4,..,28
      const float* gp = x + (size_t)(mblk + row) * EMB + k0 + c;
      v4f v = *(const v4f*)gp;
      __builtin_prefetch(gp + 32, 0, 1);    // next K-tile
      v4bf bv;
      bv[0] = cvt_bf16(v[0]); bv[1] = cvt_bf16(v[1]);
      bv[2] = cvt_bf16(v[2]); bv[3] = cvt_bf16(v[3]);
      *(v4bf*)(&ldsA[row][c]) = bv;
    }
    // ---- stage B: W[k0..+31][nblk..+127] fp32 -> bf16 LDS (transposed) ----
#pragma unroll
    for (int i = 0; i < 4; ++i) {
      int idx = tid + (i << 8);
      int kk  = idx >> 5;                   // 0..31
      int n   = (idx & 31) << 2;            // 0..124
      const float* gp = W + (size_t)(k0 + kk) * N3E + nblk + n;
      v4f v = *(const v4f*)gp;
      __builtin_prefetch(gp + (size_t)32 * N3E, 0, 1);  // next K-tile
      ldsB[n + 0][kk] = cvt_bf16(v[0]);
      ldsB[n + 1][kk] = cvt_bf16(v[1]);
      ldsB[n + 2][kk] = cvt_bf16(v[2]);
      ldsB[n + 3][kk] = cvt_bf16(v[3]);
    }
    __syncthreads();

    // ---- fragments + WMMA ----
    v16bf afr[4];
#pragma unroll
    for (int mi = 0; mi < 4; ++mi) {
      int row = wm + mi * 16 + l16;
      v8bf lo = *(const v8bf*)(&ldsA[row][half * 8]);
      v8bf hi = *(const v8bf*)(&ldsA[row][16 + half * 8]);
      afr[mi] = concat8(lo, hi);
    }
    v16bf bfr[2];
#pragma unroll
    for (int ni = 0; ni < 2; ++ni) {
      int row = wn + ni * 16 + l16;
      v8bf lo = *(const v8bf*)(&ldsB[row][half * 8]);
      v8bf hi = *(const v8bf*)(&ldsB[row][16 + half * 8]);
      bfr[ni] = concat8(lo, hi);
    }
#pragma unroll
    for (int mi = 0; mi < 4; ++mi)
#pragma unroll
      for (int ni = 0; ni < 2; ++ni)
        acc[mi][ni] = __builtin_amdgcn_wmma_f32_16x16x32_bf16(
            false, afr[mi], false, bfr[ni], (short)0, acc[mi][ni], false, false);
    __syncthreads();
  }

  // ---- epilogue: +bias, bf16, scatter to [which][head][t][hd] ----
#pragma unroll
  for (int ni = 0; ni < 2; ++ni) {
    int n_global = nblk + wn + ni * 16 + l16;
    float bv = bias[n_global];
    int which = n_global >> 10;              // 0=q 1=k 2=v
    int rem   = n_global & 1023;
    int head  = rem >> 6;
    int hd    = rem & 63;
    __bf16* base = qkv + ((size_t)which * NHEAD + head) * (size_t)(T_SEQ * HDIM) + hd;
#pragma unroll
    for (int mi = 0; mi < 4; ++mi) {
#pragma unroll
      for (int r = 0; r < 8; ++r) {
        int m = mblk + wm + mi * 16 + r + (half << 3);
        base[(size_t)m * HDIM] = cvt_bf16(acc[mi][ni][r] + bv);
      }
    }
  }
}

// ---------------------------------------------------------------------------
// Kernel 2: flash attention.  grid = (head, qblock); 128 threads = 4 waves;
// wave owns 16 query rows; streams keys in 64-wide chunks, double-buffered
// with async global->LDS copies overlapping the WMMA/softmax of the current
// chunk.
// ---------------------------------------------------------------------------
__global__ __launch_bounds__(128)
void attn_flash_bf16_wmma(const __bf16* __restrict__ qkv,
                          float* __restrict__ out) {
  __shared__ __bf16 ldsQ[64][72];           // [q][hd]
  __shared__ __bf16 ldsK[2][64][72];        // [key][hd]      (double buffered)
  __shared__ __bf16 ldsV[2][64][72];        // [hd][key] (T)  (double buffered)
  __shared__ __bf16 ldsP[4][16][72];        // per-wave P transpose buffer

  const int tid  = threadIdx.x;
  const int lane = tid & 31;
  const int wave = tid >> 5;
  const int half = lane >> 4;
  const int l16  = lane & 15;
  const int head = blockIdx.x;
  const int qblk = blockIdx.y << 6;         // 64 queries per block

  const __bf16* Qh = qkv + ((size_t)0 * NHEAD + head) * (size_t)(T_SEQ * HDIM);
  const __bf16* Kh = qkv + ((size_t)1 * NHEAD + head) * (size_t)(T_SEQ * HDIM);
  const __bf16* Vh = qkv + ((size_t)2 * NHEAD + head) * (size_t)(T_SEQ * HDIM);

  // ---- staging helpers ----
  auto stageK = [&](int buf, int kc) {      // async: ASYNCcnt-tracked
#pragma unroll
    for (int i = 0; i < 4; ++i) {
      int idx = tid + (i << 7);             // 0..511
      int row = idx >> 3;                   // 0..63
      int c   = (idx & 7) << 3;             // 0..56
      async_copy_b128(lds_off(&ldsK[buf][row][c]),
                      Kh + (size_t)(kc + row) * HDIM + c);
    }
  };
  auto stageV = [&](int buf, int kc) {      // manual transpose
#pragma unroll
    for (int i = 0; i < 4; ++i) {
      int idx = tid + (i << 7);
      int key = idx >> 3;
      int c   = (idx & 7) << 3;
      v8bf v = *(const v8bf*)(Vh + (size_t)(kc + key) * HDIM + c);
#pragma unroll
      for (int j = 0; j < 8; ++j) ldsV[buf][c + j][key] = v[j];
    }
  };

  // stage Q (async, one time) + first K/V chunk
#pragma unroll
  for (int i = 0; i < 4; ++i) {
    int idx = tid + (i << 7);
    int row = idx >> 3;
    int c   = (idx & 7) << 3;
    async_copy_b128(lds_off(&ldsQ[row][c]),
                    Qh + (size_t)(qblk + row) * HDIM + c);
  }
  stageK(0, 0);
  stageV(0, 0);
  wait_async0();
  __syncthreads();

  // Q A-fragments are loop invariant
  v16bf qfr[2];
#pragma unroll
  for (int ks = 0; ks < 2; ++ks) {
    int row = wave * 16 + l16;
    v8bf lo = *(const v8bf*)(&ldsQ[row][ks * 32 + half * 8]);
    v8bf hi = *(const v8bf*)(&ldsQ[row][ks * 32 + 16 + half * 8]);
    qfr[ks] = concat8(lo, hi);
  }

  float m_s[8], l_s[8];
  v8f O[4] = {};
#pragma unroll
  for (int r = 0; r < 8; ++r) { m_s[r] = -1.0e30f; l_s[r] = 0.0f; }

  int buf = 0;
  for (int kc = 0; kc < T_SEQ; kc += 64) {
    // ---- issue async staging of the NEXT chunk (overlaps compute) ----
    int nxt = kc + 64;
    if (nxt < T_SEQ) {
      stageK(buf ^ 1, nxt);
      stageV(buf ^ 1, nxt);
    }
    // prefetch the +2 chunk into cache (64B per thread covers 8KB per stream)
    if (kc + 128 < T_SEQ) {
      __builtin_prefetch(Kh + (size_t)(kc + 128) * HDIM + tid * 32, 0, 1);
      __builtin_prefetch(Vh + (size_t)(kc + 128) * HDIM + tid * 32, 0, 1);
    }

    // ---- S = Q K^T  (4 key sub-tiles of 16, K-dim 64 = 2 wmma steps) ----
    v8f s[4] = {};
#pragma unroll
    for (int nt = 0; nt < 4; ++nt) {
#pragma unroll
      for (int ks = 0; ks < 2; ++ks) {
        int row = nt * 16 + l16;
        v8bf lo = *(const v8bf*)(&ldsK[buf][row][ks * 32 + half * 8]);
        v8bf hi = *(const v8bf*)(&ldsK[buf][row][ks * 32 + 16 + half * 8]);
        s[nt] = __builtin_amdgcn_wmma_f32_16x16x32_bf16(
            false, qfr[ks], false, concat8(lo, hi), (short)0, s[nt], false, false);
      }
    }

    // ---- online softmax (rows live in 16-lane halves; half-wave reduce) ----
#pragma unroll
    for (int nt = 0; nt < 4; ++nt)
#pragma unroll
      for (int r = 0; r < 8; ++r) s[nt][r] *= 0.125f;   // 1/sqrt(64)

    float alpha[8];
#pragma unroll
    for (int r = 0; r < 8; ++r) {
      float mx = fmaxf(fmaxf(s[0][r], s[1][r]), fmaxf(s[2][r], s[3][r]));
#pragma unroll
      for (int msk = 1; msk < 16; msk <<= 1)
        mx = fmaxf(mx, __shfl_xor(mx, msk, 32));
      float m_new = fmaxf(m_s[r], mx);
      alpha[r] = __expf(m_s[r] - m_new);
      m_s[r] = m_new;
    }
#pragma unroll
    for (int r = 0; r < 8; ++r) {
      float rs = 0.0f;
#pragma unroll
      for (int nt = 0; nt < 4; ++nt) {
        float p = __expf(s[nt][r] - m_s[r]);
        s[nt][r] = p;
        rs += p;
      }
#pragma unroll
      for (int msk = 1; msk < 16; msk <<= 1)
        rs += __shfl_xor(rs, msk, 32);
      l_s[r] = l_s[r] * alpha[r] + rs;
#pragma unroll
      for (int ht = 0; ht < 4; ++ht) O[ht][r] *= alpha[r];
    }

    // ---- transpose P (C/D layout -> A layout) through per-wave LDS ----
#pragma unroll
    for (int nt = 0; nt < 4; ++nt)
#pragma unroll
      for (int r = 0; r < 8; ++r)
        ldsP[wave][r + (half << 3)][nt * 16 + l16] = cvt_bf16(s[nt][r]);
    // same-wave LDS ops are in-order: store->load hazard handled by hw/compiler

    // ---- O += P V  (4 hd sub-tiles of 16, K-dim 64 = 2 wmma steps) ----
#pragma unroll
    for (int ks = 0; ks < 2; ++ks) {
      v8bf plo = *(const v8bf*)(&ldsP[wave][l16][ks * 32 + half * 8]);
      v8bf phi = *(const v8bf*)(&ldsP[wave][l16][ks * 32 + 16 + half * 8]);
      v16bf pf = concat8(plo, phi);
#pragma unroll
      for (int ht = 0; ht < 4; ++ht) {
        int row = ht * 16 + l16;
        v8bf vlo = *(const v8bf*)(&ldsV[buf][row][ks * 32 + half * 8]);
        v8bf vhi = *(const v8bf*)(&ldsV[buf][row][ks * 32 + 16 + half * 8]);
        O[ht] = __builtin_amdgcn_wmma_f32_16x16x32_bf16(
            false, pf, false, concat8(vlo, vhi), (short)0, O[ht], false, false);
      }
    }

    wait_async0();        // our async copies for the next chunk are in LDS
    __syncthreads();      // everyone done reading buf / writing buf^1
    buf ^= 1;
  }

  // ---- epilogue: O / l, fp32 store to y[T, N, H] ----
#pragma unroll
  for (int ht = 0; ht < 4; ++ht) {
#pragma unroll
    for (int r = 0; r < 8; ++r) {
      int q  = qblk + wave * 16 + r + (half << 3);
      int hd = ht * 16 + l16;
      out[((size_t)q * NHEAD + head) * HDIM + hd] = O[ht][r] / l_s[r];
    }
  }
}

// ---------------------------------------------------------------------------
extern "C" void kernel_launch(void* const* d_in, const int* in_sizes, int n_in,
                              void* d_out, int out_size, void* d_ws, size_t ws_size,
                              hipStream_t stream) {
  const float* x    = (const float*)d_in[0];   // [4096, 1024]
  const float* W    = (const float*)d_in[1];   // [1024, 3072]
  const float* bqkv = (const float*)d_in[2];   // [3072]
  float* out = (float*)d_out;                  // [4096, 16, 64]
  __bf16* qkv = (__bf16*)d_ws;                 // 24 MiB: [3][16][4096][64] bf16

  (void)in_sizes; (void)n_in; (void)out_size; (void)ws_size;

  qkv_gemm_bf16_wmma<<<dim3(N3E / 128, T_SEQ / 128), 256, 0, stream>>>(x, W, bqkv, qkv);
  attn_flash_bf16_wmma<<<dim3(NHEAD, T_SEQ / 64), 128, 0, stream>>>(qkv, out);
}
